// LinkPredictor_41781441855682
// MI455X (gfx1250) — compile-verified, matrix operand order checked by reading, not run on hardware
//
#include <hip/hip_runtime.h>

typedef __attribute__((ext_vector_type(16))) __bf16 v16bf;
typedef __attribute__((ext_vector_type(16))) float  v16f;
typedef __attribute__((ext_vector_type(8)))  float  v8f;

// ---------------- degree / dinv ----------------
__global__ void k_init_deg(float* deg, int N) {
  int i = blockIdx.x * blockDim.x + threadIdx.x;
  if (i < N) deg[i] = 1.0f;                      // self-loop
}
__global__ void k_count_deg(const long long* __restrict__ dst, float* deg, long long E) {
  long long i = (long long)blockIdx.x * blockDim.x + threadIdx.x;
  if (i < E) atomicAdd(&deg[dst[i]], 1.0f);
}
__global__ void k_deg_to_dinv(float* deg, int N) {
  int i = blockIdx.x * blockDim.x + threadIdx.x;
  if (i < N) deg[i] = rsqrtf(deg[i]);
}

// Stage W[K x 128] f32 into LDS pre-swizzled to the WMMA B-register layout:
// record index ((c*8 + T)*32 + L) holds the 16 contiguous bf16 values lane L
// of N-tile T needs for K-chunk c (elements 0-7 -> k0..k0+7, 8-15 -> k0+16..k0+23,
// k0 = c*32 + (L>>4)*8; column n = T*16 + (L&15)).
__device__ __forceinline__ void stage_B(__bf16* Wlds, const float* __restrict__ W,
                                        int elems /* K*128 */, int tid) {
  for (int i = tid; i < elems; i += 256) {
    const int e = i & 15;
    const int L = (i >> 4) & 31;
    const int T = (i >> 9) & 7;
    const int c = i >> 12;
    const int k = c * 32 + (L >> 4) * 8 + (e & 7) + ((e >> 3) << 4);
    const int n = T * 16 + (L & 15);
    Wlds[i] = (__bf16)W[k * 128 + n];            // hw cvt, RNE
  }
}

// ---------------- Y[M,128] = X[M,128] @ W[128,128] via bf16 WMMA ----------------
__global__ void __launch_bounds__(256)
k_gemm128_wmma(const float* __restrict__ X, const float* __restrict__ W,
               float* __restrict__ Y, int M) {
  __shared__ __bf16 Wlds[128 * 128];             // 32 KB, B-layout swizzled
  const int tid = threadIdx.y * 32 + threadIdx.x;
  stage_B(Wlds, W, 128 * 128, tid);
  __syncthreads();

  const int lane = threadIdx.x;
  const int r    = lane & 15;                    // A row / C col within tile
  const int half = lane >> 4;                    // K-half selector per WMMA layout
  const int colT = threadIdx.y;                  // 8 waves -> 8 N-tiles (full 128 cols)
  const int row0 = blockIdx.x * 16;

  int arow = row0 + r; if (arow >= M) arow = M - 1;
  const float* xr = X + (size_t)arow * 128;
  __builtin_prefetch(xr + 128, 0, 0);            // next row -> global_prefetch_b8

  const v16bf* wb = (const v16bf*)Wlds;          // 32-byte records -> 2x ds_load_b128
  v8f acc = {};
#pragma unroll
  for (int c = 0; c < 4; ++c) {                  // K = 128, 4 chunks of 32
    const int k0 = c * 32 + half * 8;
    float4 a0 = *(const float4*)(xr + k0);
    float4 a1 = *(const float4*)(xr + k0 + 4);
    float4 a2 = *(const float4*)(xr + k0 + 16);
    float4 a3 = *(const float4*)(xr + k0 + 20);
    v16f af;
    af[0]=a0.x;  af[1]=a0.y;  af[2]=a0.z;  af[3]=a0.w;
    af[4]=a1.x;  af[5]=a1.y;  af[6]=a1.z;  af[7]=a1.w;
    af[8]=a2.x;  af[9]=a2.y;  af[10]=a2.z; af[11]=a2.w;
    af[12]=a3.x; af[13]=a3.y; af[14]=a3.z; af[15]=a3.w;
    const v16bf av = __builtin_convertvector(af, v16bf);  // v_cvt_pk_bf16_f32
    const v16bf bv = wb[(c * 8 + colT) * 32 + lane];
    acc = __builtin_amdgcn_wmma_f32_16x16x32_bf16(false, av, false, bv,
                                                  (short)0, acc, false, false);
  }
  const int n = colT * 16 + r;
#pragma unroll
  for (int v = 0; v < 8; ++v) {                  // C layout: row = v + 8*half
    const int rr = row0 + v + 8 * half;
    if (rr < M) Y[(size_t)rr * 128 + n] = acc[v];
  }
}

// ---------------- agg[dst] += h[src]*coef (32 lanes x float4 per edge) ----------------
__global__ void k_scatter(const float* __restrict__ h, const long long* __restrict__ src,
                          const long long* __restrict__ dst, const float* __restrict__ dinv,
                          float* __restrict__ agg, long long E) {
  const long long idx = (long long)blockIdx.x * blockDim.x + threadIdx.x;
  const long long e = idx >> 5;
  if (e >= E) return;
  const int q = (int)(idx & 31) * 4;
  const long long s = src[e], d = dst[e];
  const float coef = dinv[s] * dinv[d];
  const float4 v = *(const float4*)(h + (size_t)s * 128 + q);
  float* ap = agg + (size_t)d * 128 + q;
  atomicAdd(ap + 0, v.x * coef);
  atomicAdd(ap + 1, v.y * coef);
  atomicAdd(ap + 2, v.z * coef);
  atomicAdd(ap + 3, v.w * coef);
}

// ---------------- out = [relu](agg + hlin*dinv^2 + b) ----------------
__global__ void k_finalize(const float* __restrict__ agg, const float* __restrict__ hlin,
                           const float* __restrict__ dinv, const float* __restrict__ bias,
                           float* __restrict__ outp, long long total, int do_relu) {
  const long long i = (long long)blockIdx.x * blockDim.x + threadIdx.x;
  if (i >= total) return;
  const long long nidx = i >> 7;
  const int d = (int)(i & 127);
  const float dv = dinv[nidx];
  float v = agg[i] + hlin[i] * dv * dv + bias[d];
  if (do_relu) v = fmaxf(v, 0.0f);
  outp[i] = v;
}

// ---------------- fused link MLP: relu([h_s||h_d]@Wp1 + bp1) @ Wp2 + bp2 ----------------
__global__ void __launch_bounds__(256)
k_link_wmma(const float* __restrict__ h2, const long long* __restrict__ eli,
            const float* __restrict__ Wp1, const float* __restrict__ bp1,
            const float* __restrict__ Wp2, const float* __restrict__ bp2,
            float* __restrict__ out, int EL) {
  __shared__ __bf16 Wlds[256 * 128];             // 64 KB, B-layout swizzled (c = 0..7)
  const int tid = threadIdx.y * 32 + threadIdx.x;
  stage_B(Wlds, Wp1, 256 * 128, tid);
  __syncthreads();

  const int numTiles = EL >> 4;
  const int tile = blockIdx.x * 8 + threadIdx.y; // wave-uniform
  if (tile >= numTiles) return;

  const int lane = threadIdx.x, r = lane & 15, half = lane >> 4;
  const int row = tile * 16 + r;
  const long long si = eli[row], di = eli[EL + row];
  const float* fs = h2 + (size_t)si * 128;
  const float* fd = h2 + (size_t)di * 128;
  __builtin_prefetch(fd, 0, 0);

  const v16bf* wb = (const v16bf*)Wlds;
  v8f acc[8] = {};                               // 16 rows x full 128 cols
#pragma unroll
  for (int c = 0; c < 8; ++c) {                  // K = 256, chunks of 32
    const int k0 = c * 32 + half * 8;
    v16f af;
    {
      const int kk = k0;                         // 8-run never straddles the 128 split
      const float* bp = (kk < 128) ? (fs + kk) : (fd + (kk - 128));
      float4 a0 = *(const float4*)bp;
      float4 a1 = *(const float4*)(bp + 4);
      af[0]=a0.x; af[1]=a0.y; af[2]=a0.z; af[3]=a0.w;
      af[4]=a1.x; af[5]=a1.y; af[6]=a1.z; af[7]=a1.w;
    }
    {
      const int kk = k0 + 16;
      const float* bp = (kk < 128) ? (fs + kk) : (fd + (kk - 128));
      float4 a0 = *(const float4*)bp;
      float4 a1 = *(const float4*)(bp + 4);
      af[8]=a0.x;  af[9]=a0.y;  af[10]=a0.z; af[11]=a0.w;
      af[12]=a1.x; af[13]=a1.y; af[14]=a1.z; af[15]=a1.w;
    }
    const v16bf av = __builtin_convertvector(af, v16bf);
#pragma unroll
    for (int t = 0; t < 8; ++t) {                // A reused across 8 B tiles
      const v16bf bv = wb[(c * 8 + t) * 32 + lane];
      acc[t] = __builtin_amdgcn_wmma_f32_16x16x32_bf16(false, av, false, bv,
                                                       (short)0, acc[t], false, false);
    }
  }
  // epilogue: bias + ReLU + dot(Wp2), reduce across the 16-lane half-wave
  float part[8] = {0.f,0.f,0.f,0.f,0.f,0.f,0.f,0.f};
#pragma unroll
  for (int t = 0; t < 8; ++t) {
    const int n = t * 16 + r;
    const float bb = bp1[n];
    const float w2 = Wp2[n];
#pragma unroll
    for (int v = 0; v < 8; ++v) {
      float hv = acc[t][v] + bb;
      hv = fmaxf(hv, 0.0f);
      part[v] += hv * w2;
    }
  }
#pragma unroll
  for (int off = 8; off >= 1; off >>= 1) {
#pragma unroll
    for (int v = 0; v < 8; ++v) part[v] += __shfl_xor(part[v], off);
  }
  if (r == 0) {
    const float b2 = bp2[0];
#pragma unroll
    for (int v = 0; v < 8; ++v) out[tile * 16 + 8 * half + v] = part[v] + b2;
  }
}

// scalar tail for EL % 16 != 0 (not hit for EL = 200000)
__global__ void k_link_tail(const float* __restrict__ h2, const long long* __restrict__ eli,
                            const float* __restrict__ Wp1, const float* __restrict__ bp1,
                            const float* __restrict__ Wp2, const float* __restrict__ bp2,
                            float* __restrict__ out, int EL, int start) {
  const int row = start + blockIdx.x * blockDim.x + threadIdx.x;
  if (row >= EL) return;
  const float* fs = h2 + (size_t)eli[row] * 128;
  const float* fd = h2 + (size_t)eli[EL + row] * 128;
  float acc = bp2[0];
  for (int j = 0; j < 128; ++j) {
    float hv = bp1[j];
    for (int k = 0; k < 128; ++k) hv += fs[k] * Wp1[(size_t)k * 128 + j];
    for (int k = 0; k < 128; ++k) hv += fd[k] * Wp1[(size_t)(128 + k) * 128 + j];
    acc += fmaxf(hv, 0.0f) * Wp2[j];
  }
  out[row] = acc;
}

extern "C" void kernel_launch(void* const* d_in, const int* in_sizes, int n_in,
                              void* d_out, int out_size, void* d_ws, size_t ws_size,
                              hipStream_t stream) {
  const float*     x   = (const float*)d_in[0];
  const long long* ei  = (const long long*)d_in[1];
  const long long* eli = (const long long*)d_in[2];
  const float* W1  = (const float*)d_in[3];
  const float* b1  = (const float*)d_in[4];
  const float* W2  = (const float*)d_in[5];
  const float* b2  = (const float*)d_in[6];
  const float* Wp1 = (const float*)d_in[7];
  const float* bp1 = (const float*)d_in[8];
  const float* Wp2 = (const float*)d_in[9];
  const float* bp2 = (const float*)d_in[10];

  const int       N  = in_sizes[0] / 128;
  const long long E  = in_sizes[1] / 2;
  const int       EL = in_sizes[2] / 2;

  float* dinv = (float*)d_ws;                    // [N]
  float* bufA = dinv + N;                        // [N,128] linear (x@W) outputs
  float* bufB = bufA + (size_t)N * 128;          // [N,128] agg -> h (in place)
  const size_t featBytes = (size_t)N * 128 * sizeof(float);
  const long long total = (long long)N * 128;

  const int T = 256;
  const dim3 blk(32, 8);

  // symmetric-normalization coefficients
  k_init_deg   <<<(N + T - 1) / T, T, 0, stream>>>(dinv, N);
  k_count_deg  <<<(unsigned)((E + T - 1) / T), T, 0, stream>>>(ei + E, dinv, E);
  k_deg_to_dinv<<<(N + T - 1) / T, T, 0, stream>>>(dinv, N);

  // ---- GCN layer 1 (ReLU) ----
  k_gemm128_wmma<<<(N + 15) / 16, blk, 0, stream>>>(x, W1, bufA, N);
  hipMemsetAsync(bufB, 0, featBytes, stream);
  const long long sthreads = E * 32;
  k_scatter <<<(unsigned)((sthreads + T - 1) / T), T, 0, stream>>>(bufA, ei, ei + E, dinv, bufB, E);
  k_finalize<<<(unsigned)((total + T - 1) / T), T, 0, stream>>>(bufB, bufA, dinv, b1, bufB, total, 1);

  // ---- GCN layer 2 (no ReLU) ----
  k_gemm128_wmma<<<(N + 15) / 16, blk, 0, stream>>>(bufB, W2, bufA, N);
  hipMemsetAsync(bufB, 0, featBytes, stream);    // runs after gemm in stream order
  k_scatter <<<(unsigned)((sthreads + T - 1) / T), T, 0, stream>>>(bufA, ei, ei + E, dinv, bufB, E);
  k_finalize<<<(unsigned)((total + T - 1) / T), T, 0, stream>>>(bufB, bufA, dinv, b2, bufB, total, 0);

  // ---- fused link-prediction MLP ----
  const int numTiles = EL / 16;
  if (numTiles > 0)
    k_link_wmma<<<(numTiles + 7) / 8, blk, 0, stream>>>(bufB, eli, Wp1, bp1, Wp2, bp2,
                                                        (float*)d_out, EL);
  const int tail = EL - numTiles * 16;
  if (tail > 0)
    k_link_tail<<<1, 32, 0, stream>>>(bufB, eli, Wp1, bp1, Wp2, bp2,
                                      (float*)d_out, EL, numTiles * 16);
}
